// ConditionedBlockInvariantPointAttention_59115929862884
// MI455X (gfx1250) — matrix operand order repaired and checked by reading, not compiled, not run on hardware
//
#include <hip/hip_runtime.h>
#include <math.h>

typedef unsigned short u16;
typedef unsigned int   u32;
typedef __attribute__((ext_vector_type(16))) __bf16 v16bf;
typedef __attribute__((ext_vector_type(8)))  float  v8f;
typedef __attribute__((ext_vector_type(4)))  u32    u32x4;
typedef __attribute__((ext_vector_type(4)))  unsigned int tdm_g0_t;
typedef __attribute__((ext_vector_type(8)))  int          tdm_g1_t;
typedef __attribute__((ext_vector_type(4)))  int          tdm_g23_t;

static_assert(sizeof(v16bf) == 32, "v16bf must be 32 bytes");

// problem constants
constexpr int kB    = 2;
constexpr int kN    = 2048;
constexpr int kNC   = 256;
constexpr int kCS   = 384;
constexpr int kCC   = 256;
constexpr int kCZ   = 128;
constexpr int kH    = 12;
constexpr int kPQK  = 4;
constexpr int kPV   = 8;
constexpr int kBQ   = 32;
constexpr int kBK   = 128;
constexpr int kNBLK = kN / kBQ;           // 64
constexpr int kRows = kB * kN;            // 4096 token rows
constexpr int kProjC = 1152;              // q192 k192 v192 qp144 kvp432
constexpr int kFeat  = 960;               // 192 + 288 + 96 + 384
constexpr int kQKrows = kB * kNBLK * kBQ; // 4096 (b,blk,q) triples

// folded scales
#define S1_QK     0.14433756729740643f   /* sqrt(1/48) */
#define S_HW      0.13608276348795434f   /* sqrt(1/54) */
#define S_BBIAS   0.5773502691896258f    /* sqrt(1/3)  */

// ---------------------------------------------------------------- helpers
__device__ __forceinline__ u16 f2bf(float x) {
  u32 u = __float_as_uint(x);
  u32 r = (u + 0x7FFFu + ((u >> 16) & 1u)) >> 16;
  return (u16)r;
}

struct BFPair { u32x4 a, b; };

// A operand: lane holds row M=lane%16; K chunks {8h..8h+7} and {16+8h..16+8h+7}
__device__ __forceinline__ v16bf load_bfA(const u16* p, int hl) {
  BFPair t;
  t.a = *(const u32x4*)(p + 8 * hl);
  t.b = *(const u32x4*)(p + 16 + 8 * hl);
  return __builtin_bit_cast(v16bf, t);
}
// B operand (stored [N][K]): lane holds row N=lane%16; K chunks {8h..8h+7} and {16+8h..16+8h+7}
__device__ __forceinline__ v16bf load_bfB(const u16* p, int hl) {
  BFPair t;
  t.a = *(const u32x4*)(p + 8 * hl);
  t.b = *(const u32x4*)(p + 16 + 8 * hl);
  return __builtin_bit_cast(v16bf, t);
}
__device__ __forceinline__ v16bf zero_bf16() {
  BFPair t; u32x4 z = {0u, 0u, 0u, 0u}; t.a = z; t.b = z;
  return __builtin_bit_cast(v16bf, t);
}
__device__ __forceinline__ v8f wmma_bf16(v16bf a, v16bf b, v8f c) {
  return __builtin_amdgcn_wmma_f32_16x16x32_bf16(false, a, false, b, (short)0, c, false, false);
}
__device__ __forceinline__ float wave_sum(float v) {
  #pragma unroll
  for (int m = 16; m; m >>= 1) v += __shfl_xor(v, m, 32);
  return v;
}
__device__ __forceinline__ float wave_max(float v) {
  #pragma unroll
  for (int m = 16; m; m >>= 1) v = fmaxf(v, __shfl_xor(v, m, 32));
  return v;
}

// ---------------------------------------------------------------- tiny kernels
__global__ void k_convert(const float* __restrict__ src, u16* __restrict__ dst,
                          int n, float scale) {
  int i = blockIdx.x * blockDim.x + threadIdx.x;
  if (i < n) dst[i] = f2bf(src[i] * scale);
}
__global__ void k_zero16(u16* __restrict__ dst, int n) {
  int i = blockIdx.x * blockDim.x + threadIdx.x;
  if (i < n) dst[i] = 0;
}

// ---------------------------------------------------------------- cond: LN + 2 GEMMs
__global__ __launch_bounds__(256) void k_cond(
    const float* __restrict__ cond, const float* __restrict__ lnw,
    const float* __restrict__ Wg, const float* __restrict__ bg,
    const float* __restrict__ Wnb, float* __restrict__ gatepre,
    float* __restrict__ biaspre) {
  __shared__ float cn[kCC];
  __shared__ float red[8];
  int rowb = blockIdx.x;   // b*NC + nc
  int tid = threadIdx.x;
  float x = cond[(size_t)rowb * kCC + tid];
  float v = wave_sum(x);
  if ((tid & 31) == 0) red[tid >> 5] = v;
  __syncthreads();
  float s = 0.f;
  for (int i = 0; i < 8; ++i) s += red[i];
  float mu = s * (1.f / kCC);
  __syncthreads();
  float d = x - mu;
  v = wave_sum(d * d);
  if ((tid & 31) == 0) red[tid >> 5] = v;
  __syncthreads();
  s = 0.f;
  for (int i = 0; i < 8; ++i) s += red[i];
  float var = s * (1.f / kCC);
  cn[tid] = d * rsqrtf(var + 1e-5f) * lnw[tid];
  __syncthreads();
  for (int o = tid; o < kCS; o += 256) {
    float g = bg[o], nb = 0.f;
    const float* wg = Wg + (size_t)o * kCC;
    const float* wn = Wnb + (size_t)o * kCC;
    for (int c = 0; c < kCC; ++c) { float cv = cn[c]; g += wg[c] * cv; nb += wn[c] * cv; }
    gatepre[(size_t)rowb * kCS + o] = g;
    biaspre[(size_t)rowb * kCS + o] = nb;
  }
}

// ---------------------------------------------------------------- s2 = LN(s)*sig(gate)+bias
__global__ __launch_bounds__(128) void k_s2(
    const float* __restrict__ s, const float* __restrict__ gatepre,
    const float* __restrict__ biaspre, const int* __restrict__ cidx,
    const float* __restrict__ s_mask, u16* __restrict__ s2bf) {
  __shared__ float red[4];
  int row = blockIdx.x, tid = threadIdx.x;
  int b = row / kN;
  const float* sr = s + (size_t)row * kCS;
  float x0 = sr[tid], x1 = sr[tid + 128], x2 = sr[tid + 256];
  float v = wave_sum(x0 + x1 + x2);
  if ((tid & 31) == 0) red[tid >> 5] = v;
  __syncthreads();
  float mu = (red[0] + red[1] + red[2] + red[3]) * (1.f / kCS);
  __syncthreads();
  float d0 = x0 - mu, d1 = x1 - mu, d2 = x2 - mu;
  v = wave_sum(d0 * d0 + d1 * d1 + d2 * d2);
  if ((tid & 31) == 0) red[tid >> 5] = v;
  __syncthreads();
  float var = (red[0] + red[1] + red[2] + red[3]) * (1.f / kCS);
  float rs = rsqrtf(var + 1e-5f);
  int ci = cidx[row];
  float m = s_mask[row];
  const float* gp = gatepre + ((size_t)b * kNC + ci) * kCS;
  const float* bp = biaspre + ((size_t)b * kNC + ci) * kCS;
  float dd[3] = {d0, d1, d2};
  #pragma unroll
  for (int j = 0; j < 3; ++j) {
    int i = tid + 128 * j;
    float g = gp[i] * m;
    float val = dd[j] * rs * (1.f / (1.f + expf(-g))) + bp[i] * m;
    s2bf[(size_t)row * kCS + i] = f2bf(val);
  }
}

// ---------------------------------------------------------------- generic WMMA GEMM
// C(MxN) = A(MxK,bf16,row-major) * B(NxK,bf16,[N][K])^T
// Each wave: 1 M-tile x 4 N-tiles (A loaded once per k-step, reused 4x).
__global__ __launch_bounds__(256) void k_gemm(
    const u16* __restrict__ A, const u16* __restrict__ Bm,
    float* __restrict__ Cf, u16* __restrict__ Cb,
    int M, int N, int K, int njobs) {
  int wid = threadIdx.x >> 5, lane = threadIdx.x & 31;
  int l16 = lane & 15, hl = lane >> 4;
  int ngN = N >> 6;   // groups of 4 N-tiles
  for (int job = blockIdx.x * 8 + wid; job < njobs; job += gridDim.x * 8) {
    int m0 = job / ngN, g = job % ngN;
    const u16* arow = A + (size_t)(m0 * 16 + l16) * K;
    const u16* brow = Bm + (size_t)(g * 64 + l16) * K;
    v8f cs[4] = {{}, {}, {}, {}};
    for (int kk = 0; kk < K; kk += 32) {
      v16bf a = load_bfA(arow + kk, hl);
      #pragma unroll
      for (int t = 0; t < 4; ++t)
        cs[t] = wmma_bf16(a, load_bfB(brow + (size_t)t * 16 * K + kk, hl), cs[t]);
    }
    #pragma unroll
    for (int t = 0; t < 4; ++t) {
      int col = g * 64 + t * 16 + l16;
      #pragma unroll
      for (int r = 0; r < 8; ++r) {
        int rr = m0 * 16 + r + 8 * hl;
        if (Cf) Cf[(size_t)rr * N + col] = cs[t][r];
        else    Cb[(size_t)rr * N + col] = f2bf(cs[t][r]);
      }
    }
  }
}

// ---------------------------------------------------------------- per-row frames + augmented q/k
__global__ __launch_bounds__(64) void k_pts(
    const float* __restrict__ proj, const float* __restrict__ trans,
    const float* __restrict__ rots, const float* __restrict__ hwv,
    u16* __restrict__ Qaug, u16* __restrict__ Kaug,
    u16* __restrict__ vbbf, u16* __restrict__ vptsbf,
    float* __restrict__ qs2, float* __restrict__ ks2) {
  int row = blockIdx.x, tid = threadIdx.x;
  const float* pr = proj + (size_t)row * kProjC;
  for (int i = tid; i < 192; i += 64) {
    int h = i >> 4, c = i & 15;
    Qaug[(size_t)row * 384 + h * 32 + c] = f2bf(pr[i] * S1_QK);
    Kaug[(size_t)row * 384 + h * 32 + c] = f2bf(pr[192 + i]);
    vbbf[(size_t)row * 192 + i] = f2bf(pr[384 + i]);
  }
  for (int i = tid; i < 48; i += 64) {
    int h = i >> 2, j = i & 3;
    Qaug[(size_t)row * 384 + h * 32 + 28 + j] = 0;
    Kaug[(size_t)row * 384 + h * 32 + 28 + j] = 0;
  }
  if (tid < kH) {
    int h = tid;
    float R[9], t3[3];
    #pragma unroll
    for (int i = 0; i < 9; ++i) R[i] = rots[(size_t)row * 9 + i];
    #pragma unroll
    for (int i = 0; i < 3; ++i) t3[i] = trans[(size_t)row * 3 + i];
    float hw = log1pf(expf(hwv[h])) * S_HW;
    float qq = 0.f, kq = 0.f;
    #pragma unroll
    for (int p = 0; p < kPQK; ++p) {
      const float* v3 = pr + 576 + (h * kPQK + p) * 3;
      float x = v3[0], y = v3[1], z = v3[2];
      float r0 = R[0]*x + R[1]*y + R[2]*z + t3[0];
      float r1 = R[3]*x + R[4]*y + R[5]*z + t3[1];
      float r2 = R[6]*x + R[7]*y + R[8]*z + t3[2];
      qq += r0*r0 + r1*r1 + r2*r2;
      size_t o = (size_t)row * 384 + h * 32 + 16 + p * 3;
      Qaug[o] = f2bf(r0 * hw); Qaug[o+1] = f2bf(r1 * hw); Qaug[o+2] = f2bf(r2 * hw);
    }
    #pragma unroll
    for (int p = 0; p < kPQK; ++p) {
      const float* v3 = pr + 720 + (h * (kPQK + kPV) + p) * 3;
      float x = v3[0], y = v3[1], z = v3[2];
      float r0 = R[0]*x + R[1]*y + R[2]*z + t3[0];
      float r1 = R[3]*x + R[4]*y + R[5]*z + t3[1];
      float r2 = R[6]*x + R[7]*y + R[8]*z + t3[2];
      kq += r0*r0 + r1*r1 + r2*r2;
      size_t o = (size_t)row * 384 + h * 32 + 16 + p * 3;
      Kaug[o] = f2bf(r0); Kaug[o+1] = f2bf(r1); Kaug[o+2] = f2bf(r2);
    }
    #pragma unroll
    for (int j = 0; j < kPV; ++j) {
      const float* v3 = pr + 720 + (h * (kPQK + kPV) + kPQK + j) * 3;
      float x = v3[0], y = v3[1], z = v3[2];
      float r0 = R[0]*x + R[1]*y + R[2]*z + t3[0];
      float r1 = R[3]*x + R[4]*y + R[5]*z + t3[1];
      float r2 = R[6]*x + R[7]*y + R[8]*z + t3[2];
      size_t o = (size_t)row * 288 + h * 24 + j * 3;
      vptsbf[o] = f2bf(r0); vptsbf[o+1] = f2bf(r1); vptsbf[o+2] = f2bf(r2);
    }
    qs2[(size_t)row * kH + h] = -0.5f * hw * qq;
    ks2[(size_t)row * kH + h] = -0.5f * hw * kq;
  }
}

// ---------------------------------------------------------------- z stream via TDM: LN + WMMA proj
// per block: one (b,blk,q) slice of z = 128 rows x 128 cols, DMA'd to LDS by the
// Tensor Data Mover (tensor_load_to_lds, TENSORcnt), then LN'd + projected by WMMA.
constexpr int ZSM_RAW = 0;        // 128*128 f32 = 65536 B (TDM destination)
constexpr int ZSM_ZS  = 65536;    // 128*128 bf16 = 32768 B
constexpr int ZSM_WZ  = 98304;    // 48*128 bf16 = 12288 B
constexpr int ZSM_TOTAL = 110592;

__global__ __launch_bounds__(256) void k_zproj(
    const float* __restrict__ z, const float* __restrict__ lnw,
    const float* __restrict__ lnb, const u16* __restrict__ Wz,
    float* __restrict__ bb, u16* __restrict__ pzT) {
  extern __shared__ char zsmem[];
  float* zraw = (float*)(zsmem + ZSM_RAW);
  u16*   zs   = (u16*)(zsmem + ZSM_ZS);
  u16*   wzs  = (u16*)(zsmem + ZSM_WZ);
  int qidx = blockIdx.x, tid = threadIdx.x;
  int wid = tid >> 5, lane = tid & 31, l16 = lane & 15, hl = lane >> 4;

  // ---- issue TDM load of the 128x128 fp32 tile into LDS (wave 0)
  if (tid < 32) {
    unsigned long long gaddr =
        (unsigned long long)(const void*)(z + (size_t)qidx * kBK * kCZ);
    u32 lds_off = (u32)(size_t)(void*)zraw;     // low 32 bits = LDS byte offset
    // D# group0: count=1 | lds_addr | global_addr[56:0] | type=2
    tdm_g0_t g0 = { 1u, lds_off, (u32)(gaddr & 0xFFFFFFFFu),
                    (u32)((gaddr >> 32) & 0x1FFFFFFu) | (2u << 30) };
    // D# group1: data_size=2(4B); tensor_dim0=128 @bits79:48; tensor_dim1=128
    // @bits111:80; tile_dim0=128 @127:112; tile_dim1=128 @143:128;
    // tensor_dim0_stride=128 @207:160
    tdm_g1_t g1 = { (int)0x20000, (int)(128u << 16), (int)(128u << 16),
                    (int)(128u << 16), 128, 128, 0, 0 };
    tdm_g23_t g2 = { 0, 0, 0, 0 };
    tdm_g23_t g3 = { 0, 0, 0, 0 };
    tdm_g1_t  g4 = { 0, 0, 0, 0, 0, 0, 0, 0 };
    __builtin_amdgcn_tensor_load_to_lds(g0, g1, g2, g3, g4, 0);
  }
  // overlap: stage Wz while the TDM transfer is in flight
  for (int i = tid; i < 48 * 128; i += 256) wzs[i] = Wz[i];
  float lw[4], lb[4];
  #pragma unroll
  for (int j = 0; j < 4; ++j) { lw[j] = lnw[lane + 32 * j]; lb[j] = lnb[lane + 32 * j]; }
  if (tid < 32) __builtin_amdgcn_s_wait_tensorcnt(0);
  __syncthreads();

  // ---- LayerNorm rows from LDS, write bf16 tile
  for (int r = wid; r < kBK; r += 8) {
    const float* zr = zraw + r * kCZ;
    float v0 = zr[lane], v1 = zr[lane + 32], v2 = zr[lane + 64], v3 = zr[lane + 96];
    float mu = wave_sum(v0 + v1 + v2 + v3) * (1.f / kCZ);
    float d0 = v0 - mu, d1 = v1 - mu, d2 = v2 - mu, d3 = v3 - mu;
    float var = wave_sum(d0*d0 + d1*d1 + d2*d2 + d3*d3) * (1.f / kCZ);
    float rs = rsqrtf(var + 1e-5f);
    zs[r * 128 + lane]      = f2bf(d0 * rs * lw[0] + lb[0]);
    zs[r * 128 + lane + 32] = f2bf(d1 * rs * lw[1] + lb[1]);
    zs[r * 128 + lane + 64] = f2bf(d2 * rs * lw[2] + lb[2]);
    zs[r * 128 + lane + 96] = f2bf(d3 * rs * lw[3] + lb[3]);
  }
  __syncthreads();

  // ---- WMMA projection: (128 x 128) @ (48 x 128)^T
  for (int tile = wid; tile < 24; tile += 8) {    // 8 Mtiles x 3 Ntiles
    int m0 = tile / 3, n0 = tile % 3;
    const u16* arow = zs + (m0 * 16 + l16) * 128;
    const u16* brow = wzs + (n0 * 16 + l16) * 128;
    v8f c = {};
    #pragma unroll
    for (int kk = 0; kk < 128; kk += 32)
      c = wmma_bf16(load_bfA(arow + kk, hl), load_bfB(brow + kk, hl), c);
    int ncol = n0 * 16 + l16;
    #pragma unroll
    for (int r = 0; r < 8; ++r) {
      int krow = m0 * 16 + r + 8 * hl;
      if (ncol < 12)
        bb[((size_t)qidx * kBK + krow) * kH + ncol] = c[r];
      else if (ncol < 44)
        pzT[(size_t)qidx * 4096 + (size_t)(ncol - 12) * 128 + krow] = f2bf(c[r]);
    }
  }
}

// ---------------------------------------------------------------- fused block attention
constexpr int SM_A    = 0;                       // a[q][h][k]  32*12*128 bf16 = 98304B
constexpr int SM_R2   = 98304;                   // overlaid region
constexpr int SM_QA   = SM_R2;                   // 32*12*32 bf16 = 24576B
constexpr int SM_KA   = SM_R2 + 24576;           // 128*12*32 bf16 = 98304B
constexpr int SM_LOG  = SM_R2 + 24576 + 98304;   // 32*128 f32 = 16384B
constexpr int SM_VT   = SM_R2;                   // 12*16*128 bf16 = 49152B
constexpr int SM_VPT  = SM_R2 + 49152;           // 12*32*128 bf16 = 98304B
constexpr int SM_OPT  = SM_R2 + 49152 + 98304;   // 12*32*32 f32 = 49152B
constexpr int SM_MISC = SM_R2 + 196608;          // = 294912
constexpr int SM_TOTAL = SM_MISC + 10496;        // 305408 B  (<320KB/WGP)

__global__ __launch_bounds__(512, 1) void k_attn(
    const u16* __restrict__ Qaug, const u16* __restrict__ Kaug,
    const u16* __restrict__ vbbf, const u16* __restrict__ vptsbf,
    const float* __restrict__ qs2, const float* __restrict__ ks2,
    const float* __restrict__ bbp, const u16* __restrict__ pzT,
    const float* __restrict__ s_mask, const float* __restrict__ trans,
    const float* __restrict__ rots, u16* __restrict__ feats) {
  extern __shared__ char sm[];
  u16* aS    = (u16*)(sm + SM_A);
  u16* QaS   = (u16*)(sm + SM_QA);
  u16* KaS   = (u16*)(sm + SM_KA);
  float* logit = (float*)(sm + SM_LOG);
  u16* vT    = (u16*)(sm + SM_VT);
  u16* vpT   = (u16*)(sm + SM_VPT);
  float* optS = (float*)(sm + SM_OPT);
  float* qs2S = (float*)(sm + SM_MISC);   // 384
  float* ks2S = qs2S + 384;               // 1536
  float* kmS  = ks2S + 1536;              // 128
  float* qmS  = kmS + 128;                // 32
  float* qtS  = qmS + 32;                 // 96
  float* qRS  = qtS + 96;                 // 288
  int*   kiS  = (int*)(qRS + 288);        // 128

  int tid = threadIdx.x;
  int wid = tid >> 5, lane = tid & 31, l16 = lane & 15, hl = lane >> 4;
  int bid = blockIdx.x;
  int b = bid / kNBLK, blk = bid % kNBLK;
  int q0 = blk * kBQ;
  int qrowbase = b * kN + q0;
  size_t qidx0 = (size_t)bid * kBQ;

  // ---- stage q side and k-window indices
  for (int i = tid; i < kBQ * 384; i += 512)
    QaS[i] = Qaug[(size_t)(qrowbase + i / 384) * 384 + (i % 384)];
  for (int i = tid; i < 384; i += 512)
    qs2S[i] = qs2[(size_t)(qrowbase + i / kH) * kH + (i % kH)];
  for (int i = tid; i < 32; i += 512) qmS[i] = s_mask[qrowbase + i];
  for (int i = tid; i < 96; i += 512)
    qtS[i] = trans[(size_t)(qrowbase + i / 3) * 3 + i % 3];
  for (int i = tid; i < 288; i += 512)
    qRS[i] = rots[(size_t)(qrowbase + i / 9) * 9 + i % 9];
  for (int j = tid; j < kBK; j += 512) {
    int base = q0 + j - 48;
    bool val = (base >= 0) && (base < kN);
    int kr = val ? (b * kN + base) : -1;
    kiS[j] = kr;
    kmS[j] = val ? s_mask[kr] : 0.f;
  }
  __syncthreads();
  for (int i = tid; i < kBK * 384; i += 512) {
    int kr = kiS[i / 384];
    KaS[i] = (kr >= 0) ? Kaug[(size_t)kr * 384 + (i % 384)] : (u16)0;
  }
  for (int i = tid; i < kBK * kH; i += 512) {
    int kr = kiS[i / kH];
    ks2S[i] = (kr >= 0) ? ks2[(size_t)kr * kH + (i % kH)] : 0.f;
  }
  __syncthreads();

  // ---- phase 1: logits (one K=32 WMMA/tile) + softmax, per head
  int m0 = wid >> 3, n0 = wid & 7;   // 2 x 8 tile grid over (32q x 128k)
  for (int h = 0; h < kH; ++h) {
    v8f c = {};
    v16bf av = load_bfA(&QaS[((m0 * 16 + l16) * kH + h) * 32], hl);
    v16bf bv = load_bfB(&KaS[((n0 * 16 + l16) * kH + h) * 32], hl);
    c = wmma_bf16(av, bv, c);
    #pragma unroll
    for (int r = 0; r < 8; ++r) {
      int q = m0 * 16 + r + 8 * hl;
      int k = n0 * 16 + l16;
      float v = c[r]
              + bbp[((qidx0 + q) * kBK + k) * kH + h]
              + qs2S[q * kH + h] + ks2S[k * kH + h]
              + 1.0e5f * (qmS[q] * kmS[k] - 1.f);
      logit[q * 128 + k] = v;
    }
    __syncthreads();
    #pragma unroll
    for (int t = 0; t < 2; ++t) {          // wave handles 2 softmax rows
      int q = wid * 2 + t;
      const float* lr = &logit[q * 128];
      float v0 = lr[lane], v1 = lr[lane + 32], v2 = lr[lane + 64], v3 = lr[lane + 96];
      float mx = wave_max(fmaxf(fmaxf(v0, v1), fmaxf(v2, v3)));
      float e0 = expf(v0 - mx), e1 = expf(v1 - mx), e2 = expf(v2 - mx), e3 = expf(v3 - mx);
      float inv = 1.f / wave_sum(e0 + e1 + e2 + e3);
      u16* ar = &aS[(q * kH + h) * 128];
      ar[lane] = f2bf(e0 * inv); ar[lane + 32] = f2bf(e1 * inv);
      ar[lane + 64] = f2bf(e2 * inv); ar[lane + 96] = f2bf(e3 * inv);
    }
    __syncthreads();
  }

  // ---- phase 2 staging: v and v_pts transposed to [N][K] layout
  for (int i = tid; i < kBK * 192; i += 512) {
    int j = i / 192, cc = i % 192;
    int kr = kiS[j];
    vT[cc * 128 + j] = (kr >= 0) ? vbbf[(size_t)kr * 192 + cc] : (u16)0;
  }
  for (int i = tid; i < kBK * 288; i += 512) {
    int j = i / 288, cc = i % 288;
    int kr = kiS[j];
    int h = cc / 24, sl = cc % 24;
    vpT[(h * 32 + sl) * 128 + j] = (kr >= 0) ? vptsbf[(size_t)kr * 288 + cc] : (u16)0;
  }
  for (int i = tid; i < kH * 8 * 128; i += 512) {   // zero pad slots 24..31
    int h = i >> 10, rest = i & 1023;
    vpT[(h * 32 + 24) * 128 + rest] = 0;
  }
  __syncthreads();

  // ---- phase 2: o = a@v, o_pt = a@v_pts  (72 tile-jobs over 16 waves)
  for (int job = wid; job < 72; job += 16) {
    int h = job / 6, t = job % 6;
    v8f c = {};
    if (t < 2) {
      int mm = t;
      const u16* arow = &aS[((mm * 16 + l16) * kH + h) * 128];
      const u16* brow = &vT[(h * 16 + l16) * 128];
      #pragma unroll
      for (int kk = 0; kk < 128; kk += 32)
        c = wmma_bf16(load_bfA(arow + kk, hl), load_bfB(brow + kk, hl), c);
      #pragma unroll
      for (int r = 0; r < 8; ++r) {
        int q = mm * 16 + r + 8 * hl;
        feats[(size_t)(qrowbase + q) * kFeat + h * 16 + l16] = f2bf(c[r]);
      }
    } else {
      int tt = t - 2, mm = tt >> 1, nn = tt & 1;
      const u16* arow = &aS[((mm * 16 + l16) * kH + h) * 128];
      const u16* brow = &vpT[(h * 32 + nn * 16 + l16) * 128];
      #pragma unroll
      for (int kk = 0; kk < 128; kk += 32)
        c = wmma_bf16(load_bfA(arow + kk, hl), load_bfB(brow + kk, hl), c);
      #pragma unroll
      for (int r = 0; r < 8; ++r) {
        int q = mm * 16 + r + 8 * hl;
        optS[(h * 32 + q) * 32 + nn * 16 + l16] = c[r];
      }
    }
  }
  __syncthreads();

  // ---- inverse frame rotation + point norms
  for (int i = tid; i < kH * kBQ * kPV; i += 512) {
    int h = i >> 8, q = (i >> 3) & 31, pv = i & 7;
    const float* o3 = &optS[(h * 32 + q) * 32 + pv * 3];
    float x = o3[0] - qtS[q * 3], y = o3[1] - qtS[q * 3 + 1], z = o3[2] - qtS[q * 3 + 2];
    const float* R = &qRS[q * 9];
    float ox = R[0]*x + R[3]*y + R[6]*z;   // R^T
    float oy = R[1]*x + R[4]*y + R[7]*z;
    float oz = R[2]*x + R[5]*y + R[8]*z;
    u16* f = &feats[(size_t)(qrowbase + q) * kFeat];
    f[192 + h * 24 + pv * 3]     = f2bf(ox);
    f[192 + h * 24 + pv * 3 + 1] = f2bf(oy);
    f[192 + h * 24 + pv * 3 + 2] = f2bf(oz);
    f[480 + h * 8 + pv] = f2bf(sqrtf(ox*ox + oy*oy + oz*oz + 1e-8f));
  }
  __syncthreads();

  // ---- phase 3: o_pair = a[q](12x128) @ pair_z[q](128x32), per query
  for (int job = wid; job < 64; job += 16) {
    int q = job >> 1, nn = job & 1;
    size_t pzbase = (qidx0 + q) * 4096 + (size_t)(nn * 16 + l16) * 128;
    v8f c = {};
    #pragma unroll
    for (int kk = 0; kk < 128; kk += 32) {
      v16bf av;
      if (l16 < kH) av = load_bfA(&aS[(q * kH + l16) * 128 + kk], hl);
      else          av = zero_bf16();
      v16bf bv = load_bfB(&pzT[pzbase + kk], hl);
      c = wmma_bf16(av, bv, c);
    }
    #pragma unroll
    for (int r = 0; r < 8; ++r) {
      int h = r + 8 * hl;
      if (h < kH)
        feats[(size_t)(qrowbase + q) * kFeat + 576 + h * 32 + nn * 16 + l16] = f2bf(c[r]);
    }
  }
}

// ---------------------------------------------------------------- launcher
extern "C" void kernel_launch(void* const* d_in, const int* in_sizes, int n_in,
                              void* d_out, int out_size, void* d_ws, size_t ws_size,
                              hipStream_t stream) {
  (void)in_sizes; (void)n_in; (void)out_size; (void)ws_size;
  const float* s        = (const float*)d_in[0];
  const float* cond     = (const float*)d_in[1];
  const float* z        = (const float*)d_in[2];
  const float* trans    = (const float*)d_in[3];
  const float* rots     = (const float*)d_in[4];
  const float* s_mask   = (const float*)d_in[5];
  const int*   cidx     = (const int*)d_in[6];
  const float* ln_cond_w = (const float*)d_in[7];
  const float* lin_cond_W = (const float*)d_in[8];
  const float* lin_cond_b = (const float*)d_in[9];
  const float* lin_cond_nb_W = (const float*)d_in[10];
  const float* ln_z_w   = (const float*)d_in[11];
  const float* ln_z_b   = (const float*)d_in[12];
  const float* Wq       = (const float*)d_in[13];
  const float* Wk       = (const float*)d_in[14];
  const float* Wv       = (const float*)d_in[15];
  const float* Wqp      = (const float*)d_in[16];
  const float* Wkvp     = (const float*)d_in[17];
  const float* Wb       = (const float*)d_in[18];
  const float* Wdz      = (const float*)d_in[19];
  const float* head_w   = (const float*)d_in[20];
  const float* Wout     = (const float*)d_in[21];

  char* ws = (char*)d_ws;
  size_t off = 0;
  auto alloc = [&](size_t bytes) -> char* {
    char* p = ws + off;
    off += (bytes + 255) & ~(size_t)255;
    return p;
  };
  float* gatepre = (float*)alloc((size_t)kB * kNC * kCS * 4);
  float* biaspre = (float*)alloc((size_t)kB * kNC * kCS * 4);
  u16*   s2bf    = (u16*)alloc((size_t)kRows * kCS * 2);
  u16*   Wcat    = (u16*)alloc((size_t)kProjC * kCS * 2);
  u16*   Wz      = (u16*)alloc((size_t)48 * kCZ * 2);
  u16*   Woutb   = (u16*)alloc((size_t)kCS * kFeat * 2);
  float* proj    = (float*)alloc((size_t)kRows * kProjC * 4);
  u16*   Qaug    = (u16*)alloc((size_t)kRows * 384 * 2);
  u16*   Kaug    = (u16*)alloc((size_t)kRows * 384 * 2);
  u16*   vbbf    = (u16*)alloc((size_t)kRows * 192 * 2);
  u16*   vptsbf  = (u16*)alloc((size_t)kRows * 288 * 2);
  float* qs2     = (float*)alloc((size_t)kRows * kH * 4);
  float* ks2g    = (float*)alloc((size_t)kRows * kH * 4);
  float* bbw     = (float*)alloc((size_t)kQKrows * kBK * kH * 4);
  u16*   pzT     = (u16*)alloc((size_t)kQKrows * 32 * kBK * 2);
  u16*   feats   = (u16*)alloc((size_t)kRows * kFeat * 2);

  auto cg = [](int n) { return (n + 255) / 256; };

  // weight conversions (with folded scales)
  k_convert<<<cg(192 * 384), 256, 0, stream>>>(Wq,   Wcat + 0 * 384,   192 * 384, 1.f);
  k_convert<<<cg(192 * 384), 256, 0, stream>>>(Wk,   Wcat + 192 * 384, 192 * 384, 1.f);
  k_convert<<<cg(192 * 384), 256, 0, stream>>>(Wv,   Wcat + 384 * 384, 192 * 384, 1.f);
  k_convert<<<cg(144 * 384), 256, 0, stream>>>(Wqp,  Wcat + 576 * 384, 144 * 384, 1.f);
  k_convert<<<cg(432 * 384), 256, 0, stream>>>(Wkvp, Wcat + 720 * 384, 432 * 384, 1.f);
  k_convert<<<cg(12 * 128), 256, 0, stream>>>(Wb,  Wz,            12 * 128, S_BBIAS);
  k_convert<<<cg(32 * 128), 256, 0, stream>>>(Wdz, Wz + 12 * 128, 32 * 128, 1.f);
  k_zero16<<<cg(4 * 128), 256, 0, stream>>>(Wz + 44 * 128, 4 * 128);
  k_convert<<<cg(kCS * kFeat), 256, 0, stream>>>(Wout, Woutb, kCS * kFeat, 1.f);

  // conditioning + s2
  k_cond<<<kB * kNC, 256, 0, stream>>>(cond, ln_cond_w, lin_cond_W, lin_cond_b,
                                       lin_cond_nb_W, gatepre, biaspre);
  k_s2<<<kRows, 128, 0, stream>>>(s, gatepre, biaspre, cidx, s_mask, s2bf);

  // projection GEMM (WMMA): (4096 x 384) @ (1152 x 384)^T
  {
    int M = kRows, N = kProjC, K = kCS;
    int nj = (M / 16) * (N / 64);
    k_gemm<<<(nj + 7) / 8, 256, 0, stream>>>(s2bf, Wcat, proj, nullptr, M, N, K, nj);
  }

  // frames + augmented q/k
  k_pts<<<kRows, 64, 0, stream>>>(proj, trans, rots, head_w, Qaug, Kaug, vbbf,
                                  vptsbf, qs2, ks2g);

  // stream z once via TDM: LN + WMMA projection to attention bias & pair features
  (void)hipFuncSetAttribute(reinterpret_cast<const void*>(k_zproj),
                            hipFuncAttributeMaxDynamicSharedMemorySize, ZSM_TOTAL);
  k_zproj<<<kQKrows, 256, ZSM_TOTAL, stream>>>(z, ln_z_w, ln_z_b, Wz, bbw, pzT);

  // fused block attention
  (void)hipFuncSetAttribute(reinterpret_cast<const void*>(k_attn),
                            hipFuncAttributeMaxDynamicSharedMemorySize, SM_TOTAL);
  k_attn<<<kB * kNBLK, 512, SM_TOTAL, stream>>>(Qaug, Kaug, vbbf, vptsbf, qs2,
                                                ks2g, bbw, pzT, s_mask, trans,
                                                rots, feats);

  // output GEMM (WMMA): (4096 x 960) @ (384 x 960)^T -> fp32 out
  {
    int M = kRows, N = kCS, K = kFeat;
    int nj = (M / 16) * (N / 64);
    k_gemm<<<(nj + 7) / 8, 256, 0, stream>>>(feats, Woutb, (float*)d_out, nullptr,
                                             M, N, K, nj);
  }
}
